// TreeDiffEncoder_1967095021700
// MI455X (gfx1250) — matrix operand order repaired and checked by reading, not compiled, not run on hardware
//
#include <hip/hip_runtime.h>
#include <hip/hip_bf16.h>
#include <math.h>

typedef __attribute__((ext_vector_type(16))) _Float16 v16h;
typedef __attribute__((ext_vector_type(8)))  _Float16 v8h;
typedef __attribute__((ext_vector_type(8)))  float    v8f;

union F16x16 { v16h v; v8h h[2]; _Float16 e[16]; };
union F32x8  { v8f  v; float e[8]; };

#define Tn 32
#define Bn 64
#define Nn 256
#define Ln 128
#define Mn 128
#define Kn 8
#define Dn 256
#define INn 512
#define Hn 256
#define HSTR 264   // LDS h row stride in halves (528B -> 4-dword bank step, conflict-free A reads)

__device__ __forceinline__ float sigm(float x) { return 1.0f / (1.0f + expf(-x)); }

// ---------------------------------------------------------------- converts
__global__ void cvt_f32_f16(const float* __restrict__ in, _Float16* __restrict__ out, int n) {
    int i = blockIdx.x * blockDim.x + threadIdx.x;
    if (i < n) out[i] = (_Float16)in[i];
}

// ---------------------------------------------------------------- features
// One block per (t,b) row. Builds head(384), act_e(256), sub_e(256) in f16.
__global__ __launch_bounds__(256) void features_kernel(
    const float* __restrict__ cur,        // (T,B,N,D)
    const float* __restrict__ ctx_enc,    // (B,L,DACT)
    const float* __restrict__ init_enc,   // (B,N,DACT)
    const float* __restrict__ bmem,       // (B,M,D)
    const float* __restrict__ rule_mask,  // (T,B)
    const float* __restrict__ gen_mask,   // (T,B)
    const float* __restrict__ ctx_idx_mask,  // (T,B,L)
    const float* __restrict__ ctx_mask,      // (T,B)
    const float* __restrict__ init_idx_mask, // (T,B,N)
    const float* __restrict__ init_mask,     // (T,B)
    const float* __restrict__ sub_mask,      // (T,B,K)
    const float* __restrict__ op_emb,     // (4,64)
    const float* __restrict__ prod_emb,   // (200,256)
    const float* __restrict__ field_emb,  // (100,64)
    const float* __restrict__ tok_emb,    // (5000,256)
    const int* __restrict__ op_idx, const int* __restrict__ node_idx,
    const int* __restrict__ field_idx, const int* __restrict__ rule_idx,
    const int* __restrict__ gen_idx, const int* __restrict__ sub_idx, // (T,B,K)
    _Float16* __restrict__ head16,  // (TB,384)
    _Float16* __restrict__ act16,   // (TB,256)
    _Float16* __restrict__ sub16)   // (TB,256)
{
    const int row = blockIdx.x;          // t*B + b
    const int b   = row & (Bn - 1);
    const int d   = threadIdx.x;         // 0..255

    __shared__ float mL[Ln];
    __shared__ float mN[Nn];
    __shared__ float mK[Kn];
    __shared__ int   iK[Kn];
    __shared__ float sums[3];

    if (d < Ln) mL[d] = ctx_idx_mask[(size_t)row * Ln + d];
    mN[d] = init_idx_mask[(size_t)row * Nn + d];
    if (d < Kn) { mK[d] = sub_mask[(size_t)row * Kn + d]; iK[d] = sub_idx[(size_t)row * Kn + d]; }
    __syncthreads();
    if (d == 0) { float s = 0; for (int l = 0; l < Ln; ++l) s += mL[l]; sums[0] = s; }
    if (d == 1) { float s = 0; for (int n = 0; n < Nn; ++n) s += mN[n]; sums[1] = s; }
    if (d == 2) { float s = 0; for (int k = 0; k < Kn; ++k) s += mK[k]; sums[2] = s; }
    __syncthreads();
    const float msumL = sums[0], msumN = sums[1], msumK = sums[2];

    // ctx_e (masked mean over L; all-zero mask -> plain mean)
    float s0 = 0.f, s1 = 0.f;
    const float* cb = ctx_enc + (size_t)b * Ln * Dn + d;
    for (int l = 0; l < Ln; ++l) { float v = cb[(size_t)l * Dn]; s0 += v; s1 += mL[l] * v; }
    const float ctx_e = (msumL == 0.f) ? s0 * (1.f / Ln) : s1 / msumL;

    // init_e (masked mean over N)
    s0 = 0.f; s1 = 0.f;
    const float* ib = init_enc + (size_t)b * Nn * Dn + d;
    for (int n = 0; n < Nn; ++n) { float v = ib[(size_t)n * Dn]; s0 += v; s1 += mN[n] * v; }
    const float init_e = (msumN == 0.f) ? s0 * (1.f / Nn) : s1 / msumN;

    // sub_e (masked mean over K gathered memory rows; zero mask -> k=0 only)
    float ss = 0.f;
    for (int k = 0; k < Kn; ++k) {
        float w = (msumK == 0.f) ? (k == 0 ? 1.f : 0.f) : mK[k];
        ss += w * bmem[((size_t)b * Mn + iK[k]) * Dn + d];
    }
    const float sub_e = ss / ((msumK == 0.f) ? 1.f : msumK);

    const float rm = rule_mask[row], gm = gen_mask[row];
    const float cm = ctx_mask[row],  im = init_mask[row];
    const int gi = gen_idx[row], ri = rule_idx[row];
    const float gen_e  = tok_emb[(size_t)gi * Dn + d];
    const float prod_e = prod_emb[(size_t)ri * Dn + d];
    float gden = gm + cm + im; if (gden == 0.f) gden = 1.f;
    const float tok = (gen_e * gm + ctx_e * cm + init_e * im) / gden;
    const float act = prod_e * rm + tok * (1.f - rm);

    act16[(size_t)row * Dn + d] = (_Float16)act;
    sub16[(size_t)row * Dn + d] = (_Float16)sub_e;

    // head = [op(64) | field(64) | node(256)]
    const int oi = op_idx[row], fi = field_idx[row], ni = node_idx[row];
    const float* node = cur + ((size_t)row * Nn + ni) * Dn;
    float hv;
    if (d < 64)       hv = op_emb[(size_t)oi * 64 + d];
    else if (d < 128) hv = field_emb[(size_t)fi * 64 + (d - 64)];
    else              hv = node[d - 128];
    head16[(size_t)row * 384 + d] = (_Float16)hv;
    if (d < 128) head16[(size_t)row * 384 + 256 + d] = (_Float16)node[128 + d];
}

// ---------------------------------------------------------------- candidate GEMMs
// out[row, :] = A_row @ W.T + bias  (f16 WMMA, f32 accum), stored only for rows
// where operator_selection == op_id. A = [A0(:,:K0) | A1] with 32-aligned K0.
__global__ __launch_bounds__(256) void cand_gemm_kernel(
    const _Float16* __restrict__ A0, int sA0, int K0,
    const _Float16* __restrict__ A1, int sA1,
    const _Float16* __restrict__ W,  int Ktot,       // (512, Ktot) row-major
    const float* __restrict__ bias,                  // (512,)
    const int* __restrict__ opsel, int op_id,
    _Float16* __restrict__ xout)                     // (TB, 512)
{
    const int lane = threadIdx.x & 31;
    const int wv   = threadIdx.x >> 5;
    const int g    = blockIdx.x * 8 + wv;            // 4096 tiles
    const int mt   = g >> 5;                         // 128 row tiles
    const int nt   = g & 31;                         // 32 col tiles
    const int col  = lane & 15;
    const int khalf = lane >> 4;
    const int arow = mt * 16 + col;
    const int ncol = nt * 16 + col;

    F32x8 C;
    const float bv = bias[ncol];
#pragma unroll
    for (int j = 0; j < 8; ++j) C.e[j] = bv;

    for (int kk = 0; kk < Ktot; kk += 32) {
        // A fragment: 16-bit A layout, lanes 0-15 K={kk..kk+7,kk+16..23}, lanes 16-31 +8
        F16x16 a;
        const _Float16* pa = (kk < K0) ? (A0 + (size_t)arow * sA0 + kk + khalf * 8)
                                       : (A1 + (size_t)arow * sA1 + (kk - K0) + khalf * 8);
        a.h[0] = *(const v8h*)pa;
        a.h[1] = *(const v8h*)(pa + 16);
        // B fragment: same layout as A, lanes 0-15 K={kk..kk+7,kk+16..23}, lanes 16-31 +8
        const _Float16* pb = W + (size_t)ncol * Ktot + kk + khalf * 8;
        __builtin_prefetch(pb + 32, 0, 1);
        F16x16 bf;
        bf.h[0] = *(const v8h*)pb;
        bf.h[1] = *(const v8h*)(pb + 16);
        C.v = __builtin_amdgcn_wmma_f32_16x16x32_f16(false, a.v, false, bf.v,
                                                     (short)0, C.v, false, false);
    }

    const int rbase = mt * 16 + khalf * 8;           // C rows: r (lanes<16) / r+8 (lanes>=16)
#pragma unroll
    for (int r = 0; r < 8; ++r) {
        const int row = rbase + r;
        if (opsel[row] == op_id)
            xout[(size_t)row * INn + nt * 16 + col] = (_Float16)C.e[r];
    }
}

// ---------------------------------------------------------------- length-reverse
__global__ void reverse_kernel(const _Float16* __restrict__ x, const int* __restrict__ lengths,
                               _Float16* __restrict__ xr) {
    const int idx = blockIdx.x * blockDim.x + threadIdx.x;
    if (idx >= Tn * Bn * INn) return;
    const int j  = idx & (INn - 1);
    const int bt = idx >> 9;
    const int b  = bt & (Bn - 1);
    const int t  = bt >> 6;
    const int len = lengths[b];
    const int st  = (t < len) ? (len - 1 - t) : 0;
    xr[idx] = x[((size_t)st * Bn + b) * INn + j];
}

// ---------------------------------------------------------------- LSTM
// One workgroup per direction. 32 waves; each wave owns 2 tasks, a task is a
// 16x16 (batch-row x hidden-col) block with all 4 gate tiles -> cell update
// happens fully in the WMMA C-fragment registers. h (f16) lives in LDS.
__global__ __launch_bounds__(1024) void lstm_kernel(
    const _Float16* __restrict__ x_f, const _Float16* __restrict__ x_b,   // (T,B,512) f16
    const _Float16* __restrict__ Wih_f, const _Float16* __restrict__ Whh_f,
    const float* __restrict__ bias_f,
    const _Float16* __restrict__ Wih_b, const _Float16* __restrict__ Whh_b,
    const float* __restrict__ bias_b,
    const int* __restrict__ lengths,
    float* __restrict__ out)                                              // (B, 512)
{
    __shared__ __align__(16) _Float16 hsh[Bn * HSTR];
    __shared__ int len_sh[Bn];

    const int dir = blockIdx.x;
    const _Float16* x    = dir ? x_b   : x_f;
    const _Float16* Wih  = dir ? Wih_b : Wih_f;
    const _Float16* Whh  = dir ? Whh_b : Whh_f;
    const float*    bias = dir ? bias_b : bias_f;

    const int tid   = threadIdx.x;
    const int lane  = tid & 31;
    const int wv    = tid >> 5;
    const int col   = lane & 15;
    const int khalf = lane >> 4;

    for (int i = tid; i < Bn * HSTR; i += 1024) hsh[i] = (_Float16)0.f;
    if (tid < Bn) len_sh[tid] = lengths[tid];

    F32x8 c_acc[2], h_keep[2];
#pragma unroll
    for (int s = 0; s < 2; ++s)
#pragma unroll
        for (int j = 0; j < 8; ++j) { c_acc[s].e[j] = 0.f; h_keep[s].e[j] = 0.f; }

    __syncthreads();

    for (int t = 0; t < Tn; ++t) {
        F32x8 G[2][4];
        // ---- gate GEMMs (read hsh of step t-1; no writes yet)
        for (int s = 0; s < 2; ++s) {
            const int task = wv * 2 + s;
            const int m  = task >> 4;     // 4 row tiles of 16 batch rows
            const int jt = task & 15;     // 16 col tiles of 16 hidden cols
#pragma unroll
            for (int gi = 0; gi < 4; ++gi) {
                const float bv = bias[gi * Hn + jt * 16 + col];
#pragma unroll
                for (int j = 0; j < 8; ++j) G[s][gi].e[j] = bv;
            }
            const int arow = m * 16 + col;
            const _Float16* xrow = x + ((size_t)t * Bn + arow) * INn;
            // x @ Wih.T  (K = 512)
            for (int kk = 0; kk < INn; kk += 32) {
                F16x16 a;
                const _Float16* pa = xrow + kk + khalf * 8;
                a.h[0] = *(const v8h*)pa;
                a.h[1] = *(const v8h*)(pa + 16);
#pragma unroll
                for (int gi = 0; gi < 4; ++gi) {
                    const int ncol = gi * Hn + jt * 16 + col;
                    const _Float16* pb = Wih + (size_t)ncol * INn + kk + khalf * 8;
                    __builtin_prefetch(pb + 32, 0, 1);
                    F16x16 bf;
                    bf.h[0] = *(const v8h*)pb;
                    bf.h[1] = *(const v8h*)(pb + 16);
                    G[s][gi].v = __builtin_amdgcn_wmma_f32_16x16x32_f16(
                        false, a.v, false, bf.v, (short)0, G[s][gi].v, false, false);
                }
            }
            // h @ Whh.T  (K = 256), h from LDS
            for (int kk = 0; kk < Hn; kk += 32) {
                F16x16 a;
                const _Float16* ph = &hsh[arow * HSTR + kk + khalf * 8];
                a.h[0] = *(const v8h*)ph;
                a.h[1] = *(const v8h*)(ph + 16);
#pragma unroll
                for (int gi = 0; gi < 4; ++gi) {
                    const int ncol = gi * Hn + jt * 16 + col;
                    const _Float16* pb = Whh + (size_t)ncol * Hn + kk + khalf * 8;
                    F16x16 bf;
                    bf.h[0] = *(const v8h*)pb;
                    bf.h[1] = *(const v8h*)(pb + 16);
                    G[s][gi].v = __builtin_amdgcn_wmma_f32_16x16x32_f16(
                        false, a.v, false, bf.v, (short)0, G[s][gi].v, false, false);
                }
            }
        }
        __syncthreads();   // all hsh reads done before anyone writes step-t h
        // ---- cell update in C-fragment registers; masked carry for padding
        for (int s = 0; s < 2; ++s) {
            const int task = wv * 2 + s;
            const int m  = task >> 4;
            const int jt = task & 15;
            const int rbase = m * 16 + khalf * 8;
#pragma unroll
            for (int r = 0; r < 8; ++r) {
                const int brow = rbase + r;
                const bool valid = t < len_sh[brow];
                const float iv = sigm(G[s][0].e[r]);
                const float fv = sigm(G[s][1].e[r]);
                const float gv = tanhf(G[s][2].e[r]);
                const float ov = sigm(G[s][3].e[r]);
                const float cn = fv * c_acc[s].e[r] + iv * gv;
                const float hn = ov * tanhf(cn);
                if (valid) {
                    c_acc[s].e[r]  = cn;
                    h_keep[s].e[r] = hn;
                    hsh[brow * HSTR + jt * 16 + col] = (_Float16)hn;
                }
            }
        }
        __syncthreads();
    }

    // out = concat([h_f, h_b], axis=1): (B, 512)
    for (int s = 0; s < 2; ++s) {
        const int task = wv * 2 + s;
        const int m  = task >> 4;
        const int jt = task & 15;
        const int rbase = m * 16 + khalf * 8;
#pragma unroll
        for (int r = 0; r < 8; ++r) {
            const int brow = rbase + r;
            out[(size_t)brow * 512 + dir * Hn + jt * 16 + col] = h_keep[s].e[r];
        }
    }
}

// ---------------------------------------------------------------- launch
extern "C" void kernel_launch(void* const* d_in, const int* in_sizes, int n_in,
                              void* d_out, int out_size, void* d_ws, size_t ws_size,
                              hipStream_t stream) {
    (void)in_sizes; (void)n_in; (void)out_size; (void)ws_size;

    const float* cur        = (const float*)d_in[0];
    const float* ctx_enc    = (const float*)d_in[1];
    const float* init_enc   = (const float*)d_in[2];
    const float* bmem       = (const float*)d_in[3];
    const float* rule_mask  = (const float*)d_in[4];
    const float* gen_mask   = (const float*)d_in[5];
    const float* ctx_im     = (const float*)d_in[6];
    const float* ctx_mask   = (const float*)d_in[7];
    const float* init_im    = (const float*)d_in[8];
    const float* init_mask  = (const float*)d_in[9];
    const float* sub_mask   = (const float*)d_in[10];
    const float* op_emb     = (const float*)d_in[11];
    const float* prod_emb   = (const float*)d_in[12];
    const float* field_emb  = (const float*)d_in[13];
    const float* tok_emb    = (const float*)d_in[14];
    const float* stop_W     = (const float*)d_in[15];
    const float* stop_b     = (const float*)d_in[16];
    const float* del_W      = (const float*)d_in[17];
    const float* del_b      = (const float*)d_in[18];
    const float* add_W      = (const float*)d_in[19];
    const float* add_b      = (const float*)d_in[20];
    const float* sub_W      = (const float*)d_in[21];
    const float* sub_b      = (const float*)d_in[22];
    const float* Wih_f      = (const float*)d_in[23];
    const float* Whh_f      = (const float*)d_in[24];
    const float* b_f        = (const float*)d_in[25];
    const float* Wih_b      = (const float*)d_in[26];
    const float* Whh_b      = (const float*)d_in[27];
    const float* b_b        = (const float*)d_in[28];
    const int* op_idx       = (const int*)d_in[29];
    const int* node_idx     = (const int*)d_in[30];
    const int* field_idx    = (const int*)d_in[31];
    const int* rule_idx     = (const int*)d_in[32];
    const int* gen_idx      = (const int*)d_in[33];
    const int* sub_idx      = (const int*)d_in[34];
    const int* lengths      = (const int*)d_in[35];

    // workspace layout (all 16B-aligned)
    char* ws = (char*)d_ws;
    size_t o = 0;
    auto take = [&](size_t bytes) { char* p = ws + o; o += (bytes + 255) & ~(size_t)255; return p; };
    _Float16* head16  = (_Float16*)take((size_t)Tn * Bn * 384 * 2);
    _Float16* act16   = (_Float16*)take((size_t)Tn * Bn * 256 * 2);
    _Float16* sub16   = (_Float16*)take((size_t)Tn * Bn * 256 * 2);
    _Float16* x16     = (_Float16*)take((size_t)Tn * Bn * 512 * 2);
    _Float16* xrev16  = (_Float16*)take((size_t)Tn * Bn * 512 * 2);
    _Float16* stopW16 = (_Float16*)take((size_t)512 * 64 * 2);
    _Float16* delW16  = (_Float16*)take((size_t)512 * 384 * 2);
    _Float16* addW16  = (_Float16*)take((size_t)512 * 640 * 2);
    _Float16* subW16  = (_Float16*)take((size_t)512 * 640 * 2);
    _Float16* Wihf16  = (_Float16*)take((size_t)1024 * 512 * 2);
    _Float16* Whhf16  = (_Float16*)take((size_t)1024 * 256 * 2);
    _Float16* Wihb16  = (_Float16*)take((size_t)1024 * 512 * 2);
    _Float16* Whhb16  = (_Float16*)take((size_t)1024 * 256 * 2);

    auto cvt = [&](const float* src, _Float16* dst, int n) {
        cvt_f32_f16<<<(n + 255) / 256, 256, 0, stream>>>(src, dst, n);
    };
    cvt(stop_W, stopW16, 512 * 64);
    cvt(del_W,  delW16,  512 * 384);
    cvt(add_W,  addW16,  512 * 640);
    cvt(sub_W,  subW16,  512 * 640);
    cvt(Wih_f,  Wihf16,  1024 * 512);
    cvt(Whh_f,  Whhf16,  1024 * 256);
    cvt(Wih_b,  Wihb16,  1024 * 512);
    cvt(Whh_b,  Whhb16,  1024 * 256);

    features_kernel<<<Tn * Bn, 256, 0, stream>>>(
        cur, ctx_enc, init_enc, bmem, rule_mask, gen_mask, ctx_im, ctx_mask,
        init_im, init_mask, sub_mask, op_emb, prod_emb, field_emb, tok_emb,
        op_idx, node_idx, field_idx, rule_idx, gen_idx, sub_idx,
        head16, act16, sub16);

    // candidate GEMMs: only the row's selected operator writes its x vector
    const int gemm_blocks = (128 * 32) / 8;  // 4096 wave-tiles / 8 waves per block
    cand_gemm_kernel<<<gemm_blocks, 256, 0, stream>>>(
        head16, 384, 64,  (const _Float16*)nullptr, 0, stopW16, 64,  stop_b, op_idx, 0, x16);
    cand_gemm_kernel<<<gemm_blocks, 256, 0, stream>>>(
        head16, 384, 384, (const _Float16*)nullptr, 0, delW16,  384, del_b,  op_idx, 1, x16);
    cand_gemm_kernel<<<gemm_blocks, 256, 0, stream>>>(
        head16, 384, 384, act16, 256, addW16, 640, add_b, op_idx, 2, x16);
    cand_gemm_kernel<<<gemm_blocks, 256, 0, stream>>>(
        head16, 384, 384, sub16, 256, subW16, 640, sub_b, op_idx, 3, x16);

    reverse_kernel<<<(Tn * Bn * INn + 255) / 256, 256, 0, stream>>>(x16, lengths, xrev16);

    lstm_kernel<<<2, 1024, 0, stream>>>(
        x16, xrev16, Wihf16, Whhf16, b_f, Wihb16, Whhb16, b_b, lengths, (float*)d_out);
}
